// LlamaAttention_53343493816450
// MI455X (gfx1250) — matrix operand order repaired and checked by reading, not compiled, not run on hardware
//
#include <hip/hip_runtime.h>
#include <math.h>

typedef __bf16 bf16;
typedef __attribute__((ext_vector_type(16))) __bf16 v16bf;
typedef __attribute__((ext_vector_type(8)))  float  v8f;

// ---------------------------------------------------------------- fragments
// A fragment: 16x32 (MxK). lane = 16*hi + lr holds row lr.
// elems 0..7  -> K = hi*16 + 0..7 ; elems 8..15 -> K = hi*16 + 8 + 0..7
static __device__ __forceinline__ v16bf load_a_frag(const bf16* row, int hi) {
    v16bf a;
    const bf16* p1 = row + hi * 16;
    const bf16* p2 = row + hi * 16 + 8;
#pragma unroll
    for (int i = 0; i < 8; ++i) { a[i] = p1[i]; a[8 + i] = p2[i]; }
    return a;
}
// B fragment: 32x16 (KxN). lane holds column lr, K = hi*16 + 0..15 (contiguous
// in a [N,K] row-major source row).
static __device__ __forceinline__ v16bf load_b_frag(const bf16* row, int hi) {
    v16bf b;
    const bf16* p = row + hi * 16;
#pragma unroll
    for (int i = 0; i < 16; ++i) b[i] = p[i];
    return b;
}
static __device__ __forceinline__ v8f wmma_bf16(v16bf a, v16bf b, v8f c) {
    return __builtin_amdgcn_wmma_f32_16x16x32_bf16(false, a, false, b,
                                                   (short)0, c, false, false);
}
static __device__ __forceinline__ v8f zero8() {
    v8f z;
#pragma unroll
    for (int i = 0; i < 8; ++i) z[i] = 0.0f;
    return z;
}

// ---------------------------------------------------------------- packing
__global__ __launch_bounds__(256) void pack_bf16(const float* __restrict__ X,
                                                 bf16* __restrict__ Y, int n) {
    int i = blockIdx.x * 256 + threadIdx.x;
    if (i < n) Y[i] = (bf16)X[i];
}

// W [K,N] f32 -> Wt [N,K] bf16 (tiled transpose)
__global__ __launch_bounds__(256) void transpose_pack(const float* __restrict__ W,
                                                      bf16* __restrict__ Wt,
                                                      int K, int N) {
    __shared__ float tile[32][33];
    int n0 = blockIdx.x * 32, k0 = blockIdx.y * 32;
    int tx = threadIdx.x, ty = threadIdx.y;   // blockDim = (32, 8)
#pragma unroll
    for (int j = 0; j < 32; j += 8)
        tile[ty + j][tx] = W[(size_t)(k0 + ty + j) * N + n0 + tx];
    __syncthreads();
#pragma unroll
    for (int j = 0; j < 32; j += 8)
        Wt[(size_t)(n0 + ty + j) * K + k0 + tx] = (bf16)tile[tx][ty + j];
}

// ---------------------------------------------------------------- GEMM
// C[M,N] = A[M,K] @ Bt[N,K]^T.
// Block = 8 waves arranged 4(M) x 2(N): block tile 128 x 128.
// Wave tile 32(M) x 64(N): each B fragment feeds two WMMAs (8 WMMAs / K-step).
template <bool OUT_BF16>
__global__ __launch_bounds__(256) void gemm_xw(const bf16* __restrict__ A,
                                               const bf16* __restrict__ Bt,
                                               void* __restrict__ Cout,
                                               int M, int N, int K) {
    const int lane = threadIdx.x & 31;
    const int w    = threadIdx.x >> 5;
    const int lr   = lane & 15, hi = lane >> 4;
    const int mw   = w & 3, nw = w >> 2;
    const int m0   = blockIdx.y * 128 + mw * 32;
    const int n0   = blockIdx.x * 128 + nw * 64;

    v8f acc[2][4];
#pragma unroll
    for (int g = 0; g < 2; ++g)
#pragma unroll
        for (int t = 0; t < 4; ++t) acc[g][t] = zero8();

    const bf16* arow0 = A  + (size_t)(m0 + lr) * K;
    const bf16* arow1 = A  + (size_t)(m0 + 16 + lr) * K;
    const bf16* brow  = Bt + (size_t)(n0 + lr) * K;

    for (int kk = 0; kk < K; kk += 32) {
        if (kk + 128 < K) {                 // pull the K-stream into cache early
            __builtin_prefetch(arow0 + kk + 128, 0, 3);
            __builtin_prefetch(arow1 + kk + 128, 0, 3);
            __builtin_prefetch(brow + kk + 128, 0, 3);
        }
        v16bf a0 = load_a_frag(arow0 + kk, hi);
        v16bf a1 = load_a_frag(arow1 + kk, hi);
#pragma unroll
        for (int t = 0; t < 4; ++t) {
            v16bf b = load_b_frag(brow + (size_t)t * 16 * K + kk, hi);
            acc[0][t] = wmma_bf16(a0, b, acc[0][t]);
            acc[1][t] = wmma_bf16(a1, b, acc[1][t]);
        }
    }
#pragma unroll
    for (int g = 0; g < 2; ++g)
#pragma unroll
        for (int t = 0; t < 4; ++t)
#pragma unroll
            for (int r = 0; r < 8; ++r) {
                int row = m0 + g * 16 + r + 8 * hi;
                int col = n0 + t * 16 + lr;
                if (OUT_BF16)
                    ((bf16*)Cout)[(size_t)row * N + col] = (bf16)acc[g][t][r];
                else
                    ((float*)Cout)[(size_t)row * N + col] = acc[g][t][r];
            }
}

// ---------------------------------------------------------------- RoPE + relayout
// qraw/kraw/vraw: [B,T,H,128] bf16.  Qo/Ko: [B,H,T,128].  Vt: [B,H,128,T].
// Q is pre-scaled by 1/sqrt(Hd) so attention skips the scores scale.
__global__ __launch_bounds__(256) void rope_pack(const bf16* __restrict__ qr,
                                                 const bf16* __restrict__ kr,
                                                 const bf16* __restrict__ vr,
                                                 bf16* __restrict__ Qo,
                                                 bf16* __restrict__ Ko,
                                                 bf16* __restrict__ Vt, int T) {
    int i = blockIdx.x * 256 + threadIdx.x;          // one thread per (b,t,h,d<64)
    int total = 2 * T * 16 * 64;
    if (i >= total) return;
    int d = i & 63;
    int h = (i >> 6) & 15;
    int t = (i >> 10) & (T - 1);                     // T is a power of two
    int b = i / (T << 10);

    size_t bi = ((size_t)(b * T + t) * 16 + h) * 128;
    float invf = __expf(-(float)d * (9.2103403719761836f / 64.0f)); // 10000^(-d/64)
    float ang  = (float)t * invf;
    float sv, cv;
    sincosf(ang, &sv, &cv);

    const float qs = 0.08838834764831845f;           // 1/sqrt(128) folded into Q
    float q1 = (float)qr[bi + d], q2 = (float)qr[bi + 64 + d];
    float k1 = (float)kr[bi + d], k2 = (float)kr[bi + 64 + d];
    size_t oi = ((size_t)(b * 16 + h) * T + t) * 128;
    Qo[oi + d]      = (bf16)((q1 * cv - q2 * sv) * qs);
    Qo[oi + 64 + d] = (bf16)((q2 * cv + q1 * sv) * qs);
    Ko[oi + d]      = (bf16)(k1 * cv - k2 * sv);
    Ko[oi + 64 + d] = (bf16)(k2 * cv + k1 * sv);

    size_t vi = ((size_t)(b * 16 + h) * 128 + d) * T + t;
    Vt[vi]                 = vr[bi + d];
    Vt[vi + (size_t)64 * T] = vr[bi + 64 + d];
}

// ---------------------------------------------------------------- flash attention
// Q,K: [B,H,T,128] bf16 ; Vt: [B,H,128,T] bf16 ; O: [B,T,2048] bf16.
// 8 waves/block; wave owns a 16-row query tile; keys streamed in blocks of 32.
__global__ __launch_bounds__(256) void attn_fused(const bf16* __restrict__ Q,
                                                  const bf16* __restrict__ Kc,
                                                  const bf16* __restrict__ Vt,
                                                  bf16* __restrict__ O, int T) {
    __shared__ bf16 plds[8][16][32];                 // per-wave P-tile scratch
    const int lane = threadIdx.x & 31, w = threadIdx.x >> 5;
    const int lr = lane & 15, hi = lane >> 4;
    const int h = blockIdx.y, b = blockIdx.z;
    const int q0 = blockIdx.x * 128 + w * 16;

    const bf16* qh = Q  + (size_t)(b * 16 + h) * T * 128;
    const bf16* kh = Kc + (size_t)(b * 16 + h) * T * 128;
    const bf16* vh = Vt + (size_t)(b * 16 + h) * 128 * T;

    v16bf qa[4];
    const bf16* qrow = qh + (size_t)(q0 + lr) * 128;
#pragma unroll
    for (int s = 0; s < 4; ++s) qa[s] = load_a_frag(qrow + s * 32, hi);

    v8f acc[8];
#pragma unroll
    for (int n = 0; n < 8; ++n) acc[n] = zero8();
    float m_run[8], l_run[8];
#pragma unroll
    for (int r = 0; r < 8; ++r) { m_run[r] = -1e30f; l_run[r] = 0.0f; }

    const int kend = q0 + 16;                        // causal upper bound

    for (int kb = 0; kb < kend; kb += 32) {
        if (kb + 32 < kend) {                        // prefetch next key block
            __builtin_prefetch(kh + (size_t)(kb + 32 + lr) * 128, 0, 3);
            __builtin_prefetch(kh + (size_t)(kb + 48 + lr) * 128, 0, 3);
            __builtin_prefetch(vh + (size_t)lane * 8 * T + kb + 32, 0, 3);
        }
        v8f s0 = zero8(), s1 = zero8();
        const bf16* krow0 = kh + (size_t)(kb + lr) * 128;
        const bf16* krow1 = kh + (size_t)(kb + 16 + lr) * 128;
#pragma unroll
        for (int s = 0; s < 4; ++s) {
            v16bf b0 = load_b_frag(krow0 + s * 32, hi);   // B = K^T (Hd x keys)
            v16bf b1 = load_b_frag(krow1 + s * 32, hi);
            s0 = wmma_bf16(qa[s], b0, s0);
            s1 = wmma_bf16(qa[s], b1, s1);
        }
#pragma unroll
        for (int r = 0; r < 8; ++r) {
            int gq = q0 + r + 8 * hi;
            float x0 = s0[r]; if (kb + lr > gq)      x0 = -1e30f;
            float x1 = s1[r]; if (kb + 16 + lr > gq) x1 = -1e30f;
            float rm = fmaxf(x0, x1);                 // row lives in 16 lanes
            rm = fmaxf(rm, __shfl_xor(rm, 1));
            rm = fmaxf(rm, __shfl_xor(rm, 2));
            rm = fmaxf(rm, __shfl_xor(rm, 4));
            rm = fmaxf(rm, __shfl_xor(rm, 8));
            float mnew  = fmaxf(m_run[r], rm);
            float alpha = __expf(m_run[r] - mnew);
            float p0 = __expf(x0 - mnew);
            float p1 = __expf(x1 - mnew);
            float ps = p0 + p1;
            ps += __shfl_xor(ps, 1);
            ps += __shfl_xor(ps, 2);
            ps += __shfl_xor(ps, 4);
            ps += __shfl_xor(ps, 8);
            l_run[r] = l_run[r] * alpha + ps;
            m_run[r] = mnew;
#pragma unroll
            for (int n = 0; n < 8; ++n) acc[n][r] *= alpha;
            int row = r + 8 * hi;                     // C-layout -> LDS (row, key)
            plds[w][row][lr]      = (bf16)p0;
            plds[w][row][16 + lr] = (bf16)p1;
        }
        // wave-local store->load ordering (waves have different trip counts,
        // so a block barrier is illegal here)
        asm volatile("s_wait_dscnt 0x0" ::: "memory");
        v16bf pa = load_a_frag(&plds[w][lr][0], hi);  // reload in A layout
#pragma unroll
        for (int n = 0; n < 8; ++n) {
            v16bf vb = load_b_frag(vh + (size_t)(n * 16 + lr) * T + kb, hi);
            acc[n] = wmma_bf16(pa, vb, acc[n]);
        }
    }

#pragma unroll
    for (int r = 0; r < 8; ++r) {
        float inv = 1.0f / l_run[r];
        int gq = q0 + r + 8 * hi;
        bf16* orow = O + (size_t)(b * T + gq) * 2048 + h * 128;
#pragma unroll
        for (int n = 0; n < 8; ++n) orow[n * 16 + lr] = (bf16)(acc[n][r] * inv);
    }
}

// ---------------------------------------------------------------- launch
extern "C" void kernel_launch(void* const* d_in, const int* in_sizes, int n_in,
                              void* d_out, int out_size, void* d_ws, size_t ws_size,
                              hipStream_t stream) {
    const float* x  = (const float*)d_in[0];
    const float* wq = (const float*)d_in[1];
    const float* wk = (const float*)d_in[2];
    const float* wv = (const float*)d_in[3];
    const float* wo = (const float*)d_in[4];
    // d_in[5] = mask: causal, handled analytically.

    const int D = 2048, B = 2;
    const int T = in_sizes[0] / (B * D);   // 4096
    const int M = B * T;                   // 8192

    char* ws = (char*)d_ws;
    size_t off = 0;
    auto carve = [&](size_t bytes) -> void* {
        void* r = ws + off;
        off += (bytes + 255) & ~(size_t)255;
        return r;
    };
    bf16* xb   = (bf16*)carve((size_t)M * D * 2);
    bf16* wqt  = (bf16*)carve((size_t)D * D * 2);
    bf16* wkt  = (bf16*)carve((size_t)D * D * 2);
    bf16* wvt  = (bf16*)carve((size_t)D * D * 2);
    bf16* wot  = (bf16*)carve((size_t)D * D * 2);
    bf16* qraw = (bf16*)carve((size_t)M * D * 2);
    bf16* kraw = (bf16*)carve((size_t)M * D * 2);
    bf16* vraw = (bf16*)carve((size_t)M * D * 2);
    bf16* Qo   = (bf16*)carve((size_t)M * D * 2);
    bf16* Ko   = (bf16*)carve((size_t)M * D * 2);
    bf16* Vt   = (bf16*)carve((size_t)M * D * 2);
    bf16* aob  = xb;                       // xb is dead after the QKV GEMMs

    int n = M * D;
    pack_bf16<<<(n + 255) / 256, 256, 0, stream>>>(x, xb, n);

    dim3 tb(32, 8), tg(D / 32, D / 32);
    transpose_pack<<<tg, tb, 0, stream>>>(wq, wqt, D, D);
    transpose_pack<<<tg, tb, 0, stream>>>(wk, wkt, D, D);
    transpose_pack<<<tg, tb, 0, stream>>>(wv, wvt, D, D);
    transpose_pack<<<tg, tb, 0, stream>>>(wo, wot, D, D);

    dim3 gg(D / 128, M / 128);
    gemm_xw<true><<<gg, 256, 0, stream>>>(xb, wqt, qraw, M, D, D);
    gemm_xw<true><<<gg, 256, 0, stream>>>(xb, wkt, kraw, M, D, D);
    gemm_xw<true><<<gg, 256, 0, stream>>>(xb, wvt, vraw, M, D, D);

    int rp = B * T * 16 * 64;
    rope_pack<<<(rp + 255) / 256, 256, 0, stream>>>(qraw, kraw, vraw, Qo, Ko, Vt, T);

    attn_fused<<<dim3(T / 128, 16, B), 256, 0, stream>>>(Qo, Ko, Vt, aob, T);

    gemm_xw<false><<<gg, 256, 0, stream>>>(aob, wot, d_out, M, D, D);
}
